// AttentionFusionBlock_32719060861240
// MI455X (gfx1250) — matrix operand order repaired and checked by reading, not compiled, not run on hardware
//
#include <hip/hip_runtime.h>

// ---------------------------------------------------------------------------
// CDNA5 (gfx1250, wave32) implementation of the 4-stage windowed attention
// block. All GEMMs and attention matmuls run on v_wmma_f32_16x16x32_bf16.
// GEMM tiles are fed by the Tensor Data Mover (tensor_load_to_lds) when the
// toolchain exposes the builtin, with s_wait_tensorcnt synchronization.
// ---------------------------------------------------------------------------

typedef __bf16 bf16;
typedef __attribute__((ext_vector_type(16))) __bf16 v16bf;
typedef __attribute__((ext_vector_type(8)))  float  v8f;

#if __has_builtin(__builtin_amdgcn_tensor_load_to_lds) && \
    __has_builtin(__builtin_amdgcn_s_wait_tensorcnt)
#define USE_TDM 1
typedef unsigned __attribute__((ext_vector_type(4))) u32x4;
typedef int      __attribute__((ext_vector_type(4))) i32x4;
typedef int      __attribute__((ext_vector_type(8))) i32x8;
#endif

#define DEV static __device__ __forceinline__

DEV bf16 f2bf(float f) {
  unsigned u = __float_as_uint(f);
  unsigned r = (u + 0x7FFFu + ((u >> 16) & 1u)) >> 16;   // round-to-nearest-even
  unsigned short s = (unsigned short)r;
  return __builtin_bit_cast(bf16, s);
}
DEV float bf2f(bf16 b) {
  unsigned short s = __builtin_bit_cast(unsigned short, b);
  return __uint_as_float(((unsigned)s) << 16);
}
DEV v8f zero8() {
  v8f z;
#pragma unroll
  for (int i = 0; i < 8; ++i) z[i] = 0.0f;
  return z;
}

// A-matrix fragment, 16x32 bf16, row-major source with leading dim `ld`.
// ISA 7.12.2: lane = row (M), per-lane K pairs: elems 0..7 -> K=8g+0..7,
// elems 8..15 -> K=16+8g+0..7 (g = lane>>4).
DEV v16bf load_a_frag(const bf16* src, int ld) {
  int lane = threadIdx.x & 31;
  const bf16* r = src + (lane & 15) * ld;
  int g8 = (lane >> 4) * 8;
  v16bf a;
#pragma unroll
  for (int e = 0; e < 8; ++e) a[e] = r[g8 + e];
#pragma unroll
  for (int e = 0; e < 8; ++e) a[8 + e] = r[16 + g8 + e];
  return a;
}

// B-matrix fragment, 32x16 bf16. Source is row-major with one row per output
// column (i.e. W[N,K] layout): lane&15 = column, elems 0..7 -> K=8g+0..7,
// elems 8..15 -> K=16+8g+0..7 (g = lane>>4).
DEV v16bf load_b_frag(const bf16* src, int ld) {
  int lane = threadIdx.x & 31;
  const bf16* r = src + (lane & 15) * ld + (lane >> 4) * 8;
  v16bf b;
#pragma unroll
  for (int e = 0; e < 8; ++e) b[e] = r[e];
#pragma unroll
  for (int e = 0; e < 8; ++e) b[8 + e] = r[16 + e];
  return b;
}

DEV v8f wmma_bf16(v16bf a, v16bf b, v8f c) {
  // D = A(16x32) * B(32x16) + C, f32 accum
  return __builtin_amdgcn_wmma_f32_16x16x32_bf16(false, a, false, b,
                                                 (short)0, c, false, false);
}

#ifdef USE_TDM
// Issue one TDM 2-D tile load: `rows` rows of 16 dwords (32 bf16) each, row
// stride `stride_dw` dwords, destination LDS with 4 pad dwords after every 16
// (=> 40-bf16 padded LDS rows). Wave-level op; caller must be a single wave
// and follow with s_wait_tensorcnt + workgroup barrier.
DEV void tdm_load_tile(const void* gptr, unsigned lds_byte_addr,
                       int rows, int stride_dw) {
  unsigned long long ga = (unsigned long long)(uintptr_t)gptr;
  u32x4 g0;
  g0[0] = 1u;                                            // count=1, user D#
  g0[1] = lds_byte_addr;                                 // LDS dest (bytes)
  g0[2] = (unsigned)(ga & 0xFFFFFFFFu);                  // global_addr lo
  g0[3] = (unsigned)((ga >> 32) & 0x01FFFFFFu) | 0x80000000u; // hi + type=2
  i32x8 g1;
  // data_size=4B (2), pad_enable, pad_interval=3 (16 dwords), pad_amount=3 (4)
  unsigned w0 = (2u << 16) | (1u << 20) | (3u << 22) | (3u << 25);
  unsigned td0 = 16u;                                    // tensor_dim0 (dwords)
  unsigned td1 = (unsigned)rows;                         // tensor_dim1 (rows)
  g1[0] = (int)w0;
  g1[1] = (int)((td0 & 0xFFFFu) << 16);                  // tensor_dim0 lo16
  g1[2] = (int)((td0 >> 16) | ((td1 & 0xFFFFu) << 16));  // dim0 hi / dim1 lo
  g1[3] = (int)((td1 >> 16) | (16u << 16));              // dim1 hi / tile_dim0
  g1[4] = (int)(td1 & 0xFFFFu);                          // tile_dim1 (tile_dim2=0)
  g1[5] = (int)(unsigned)stride_dw;                      // tensor_dim0_stride lo32
  g1[6] = 0;                                             // stride hi / dim1_stride lo
  g1[7] = 0;
  i32x4 z4 = {0, 0, 0, 0};
#if __clang_major__ >= 23
  i32x8 z8 = {0, 0, 0, 0, 0, 0, 0, 0};
  __builtin_amdgcn_tensor_load_to_lds(g0, g1, z4, z4, z8, 0);
#else
  __builtin_amdgcn_tensor_load_to_lds(g0, g1, z4, z4, 0);
#endif
}
#endif  // USE_TDM

// ---------------------------------------------------------------------------
// Layout constants
// ---------------------------------------------------------------------------
#define HDIM 192
#define WDIM 192
#define BV   2
#define NWH  24
#define CDIM 256
#define TTOK (BV * HDIM * WDIM)      // 73728 tokens
#define NWIN (BV * NWH * NWH)        // 1152 windows

enum { EPI_STORE = 0, EPI_GELU = 1, EPI_RESID = 2 };

// ---------------------------------------------------------------------------
// f32 -> bf16 converter (weights)
// ---------------------------------------------------------------------------
__global__ void k_cvt(const float* __restrict__ in, bf16* __restrict__ out, int n) {
#pragma unroll
  for (int t = 0; t < 4; ++t) {
    int idx = blockIdx.x * 1024 + t * 256 + threadIdx.x;
    if (idx < n) out[idx] = f2bf(in[idx]);
  }
}

// NCHW f32 -> [T, C] bf16 tokens
__global__ void k_to_tok(const float* __restrict__ src, bf16* __restrict__ tok) {
  size_t idx = (size_t)blockIdx.x * 256 + threadIdx.x;   // over T*C
  int t = (int)(idx >> 8);
  int c = (int)(idx & 255);
  int b = t / (HDIM * WDIM);
  int hw = t - b * HDIM * WDIM;
  tok[idx] = f2bf(src[((size_t)(b * CDIM + c)) * (HDIM * WDIM) + hw]);
}

// [T, C] bf16 tokens -> NCHW f32
__global__ void k_from_tok(const bf16* __restrict__ tok, float* __restrict__ dst) {
  size_t idx = (size_t)blockIdx.x * 256 + threadIdx.x;   // over B*C*H*W
  int b = (int)(idx / ((size_t)CDIM * HDIM * WDIM));
  int rem = (int)(idx - (size_t)b * CDIM * HDIM * WDIM);
  int c = rem / (HDIM * WDIM);
  int hw = rem - c * (HDIM * WDIM);
  dst[idx] = bf2f(tok[((size_t)(b * HDIM * WDIM + hw)) * CDIM + c]);
}

// ---------------------------------------------------------------------------
// GEMM: out[M, :] = A[M,K](bf16) * W[N,K]^T(bf16) + bias, epilogue selected
// at compile time (store / exact GELU / residual add). Block = 64x256 output
// tile, 8 waves (2 along M x 4 along N), each wave 32x64 via 2x4 WMMA
// fragments. K tiled by 32 through LDS, tiles DMA'd by the TDM when present.
// ---------------------------------------------------------------------------
template <int EPI>
__global__ __launch_bounds__(256) void k_gemm(
    const bf16* __restrict__ A, const bf16* __restrict__ W,
    const float* __restrict__ bias, bf16* __restrict__ out,
    const bf16* __restrict__ resid, int K, int ldOut, int colOff)
{
  __shared__ __align__(16) bf16 As[64][40];
  __shared__ __align__(16) bf16 Bs[256][40];

  const int tid  = threadIdx.x;
  const int lane = tid & 31;
  const int wv   = tid >> 5;
  const int wm   = wv & 1;          // wave row (0..1)
  const int wn   = wv >> 1;         // wave col (0..3)
  const size_t mBase = (size_t)blockIdx.x * 64;
  const int nMatBase = blockIdx.y * 256;

  v8f acc[2][4];
#pragma unroll
  for (int im = 0; im < 2; ++im)
#pragma unroll
    for (int jn = 0; jn < 4; ++jn) acc[im][jn] = zero8();

#ifdef USE_TDM
  const unsigned ldsA = (unsigned)(uintptr_t)&As[0][0];
  const unsigned ldsB = (unsigned)(uintptr_t)&Bs[0][0];
#endif

  for (int k0 = 0; k0 < K; k0 += 32) {
#ifdef USE_TDM
    if (wv == 0) {
      // A tile: 64 rows x 32 bf16 ; B tile: 256 weight rows x 32 bf16
      tdm_load_tile(A + mBase * (size_t)K + k0, ldsA, 64, K >> 1);
      tdm_load_tile(W + (size_t)nMatBase * K + k0, ldsB, 256, K >> 1);
      __builtin_amdgcn_s_wait_tensorcnt(0);
    }
    __syncthreads();
#else
    {
      const int arow = tid >> 2, achunk = tid & 3;
      *(uint4*)&As[arow][achunk * 8] =
          *(const uint4*)&A[(mBase + arow) * (size_t)K + k0 + achunk * 8];
      const uint4* bs = (const uint4*)&W[((size_t)(nMatBase + tid)) * K + k0];
      uint4* bd = (uint4*)&Bs[tid][0];
      bd[0] = bs[0]; bd[1] = bs[1]; bd[2] = bs[2]; bd[3] = bs[3];
    }
    __syncthreads();
#endif

    v16bf af[2], bw[4];
    af[0] = load_a_frag(&As[32 * wm][0], 40);
    af[1] = load_a_frag(&As[32 * wm + 16][0], 40);
#pragma unroll
    for (int jn = 0; jn < 4; ++jn)
      bw[jn] = load_b_frag(&Bs[64 * wn + 16 * jn][0], 40);
#pragma unroll
    for (int im = 0; im < 2; ++im)
#pragma unroll
      for (int jn = 0; jn < 4; ++jn)
        acc[im][jn] = wmma_bf16(af[im], bw[jn], acc[im][jn]);
    __syncthreads();
  }

  // Epilogue. C-matrix layout: VGPR r holds M=r (lanes 0-15) / M=r+8 (16-31),
  // lane&15 = N within the 16-wide fragment.
  const int g = lane >> 4, nl = lane & 15;
#pragma unroll
  for (int im = 0; im < 2; ++im) {
#pragma unroll
    for (int jn = 0; jn < 4; ++jn) {
#pragma unroll
      for (int r = 0; r < 8; ++r) {
        size_t m = mBase + 32 * wm + 16 * im + r + 8 * g;
        int nloc = 64 * wn + 16 * jn + nl;
        int nmat = nMatBase + nloc;
        float v = acc[im][jn][r] + bias[nmat];
        if (EPI == EPI_GELU) {
          v = 0.5f * v * (1.0f + erff(v * 0.70710678118654752f));
        } else if (EPI == EPI_RESID) {
          v += bf2f(resid[m * CDIM + nloc]);
        }
        out[m * (size_t)ldOut + colOff + nmat] = f2bf(v);
      }
    }
  }
}

// ---------------------------------------------------------------------------
// Fused windowed attention: one (window, head-pair) per 64-thread block
// (2 waves, one head each). Q*K^T -> bias/mask -> softmax -> P*V, all WMMA.
// qbuf/kvbuf are [T, 768] bf16 with q at col 0, k at 256, v at 512.
// off = roll shift (0 or 4); masked = Swin shift mask enable.
// ---------------------------------------------------------------------------
__global__ __launch_bounds__(64) void k_attn(
    const bf16* __restrict__ qbuf, const bf16* __restrict__ kvbuf,
    const float* __restrict__ rpb, bf16* __restrict__ outp,
    int off, int masked)
{
  __shared__ __align__(16) bf16 qs [2][64][32];
  __shared__ __align__(16) bf16 ksm[2][64][32];
  __shared__ __align__(16) bf16 vt [2][32][64];   // V transposed: [dim][pos]
  __shared__ __align__(16) bf16 ps [2][64][64];   // probabilities
  __shared__ int   toks[64];
  __shared__ float rpbs[225 * 8];

  const int tid  = threadIdx.x;
  const int lane = tid & 31;
  const int wv   = tid >> 5;
  const int head = blockIdx.y * 2 + wv;

  const int wid   = blockIdx.x;
  const int b     = wid / (NWH * NWH);
  const int r2    = wid - b * NWH * NWH;
  const int win_h = r2 / NWH;
  const int win_w = r2 - win_h * NWH;

  {  // token index for each of the 64 window positions (gather with roll)
    int r = tid >> 3, c = tid & 7;
    int oh = (win_h * 8 + r + off) % HDIM;
    int ow = (win_w * 8 + c + off) % WDIM;
    toks[tid] = (b * HDIM + oh) * WDIM + ow;
  }
  for (int i = tid; i < 225 * 8; i += 64) rpbs[i] = rpb[i];
  __syncthreads();

  // Stage Q, K (row-major) and V (transposed) for both heads of this block.
  for (int rr = tid; rr < 128; rr += 64) {
    int hh = rr >> 6, p = rr & 63;
    int hv = blockIdx.y * 2 + hh;
    size_t base = (size_t)toks[p] * 768;
    const uint4* q4 = (const uint4*)(qbuf + base + hv * 32);
    ((uint4*)&qs[hh][p][0])[0] = q4[0];
    ((uint4*)&qs[hh][p][0])[1] = q4[1];
    const uint4* k4 = (const uint4*)(kvbuf + base + 256 + hv * 32);
    ((uint4*)&ksm[hh][p][0])[0] = k4[0];
    ((uint4*)&ksm[hh][p][0])[1] = k4[1];
    const bf16* vsrc = kvbuf + base + 512 + hv * 32;
#pragma unroll
    for (int d = 0; d < 32; ++d) vt[hh][d][p] = vsrc[d];
  }
  __syncthreads();

  // ---- scores: S = Q(64x32) * K^T(32x64), 4x4 fragments, single K-step ----
  v16bf aq[4], bk[4];
#pragma unroll
  for (int i = 0; i < 4; ++i) aq[i] = load_a_frag(&qs[wv][16 * i][0], 32);
#pragma unroll
  for (int j = 0; j < 4; ++j) bk[j] = load_b_frag(&ksm[wv][16 * j][0], 32);
  v8f s[4][4];
#pragma unroll
  for (int i = 0; i < 4; ++i)
#pragma unroll
    for (int j = 0; j < 4; ++j)
      s[i][j] = wmma_bf16(aq[i], bk[j], zero8());

  // ---- bias + mask + softmax (per score row, 16-lane group reductions) ----
  const float scale = 0.17677669529663687f;  // 1/sqrt(32)
  const int g = lane >> 4, nl = lane & 15;
#pragma unroll
  for (int i = 0; i < 4; ++i) {
#pragma unroll
    for (int r = 0; r < 8; ++r) {
      int m  = 16 * i + r + 8 * g;
      int rm = m >> 3, cm = m & 7;
      int regm = 0;
      if (masked) {
        int rh = (win_h == NWH - 1) ? (rm < 4 ? 1 : 2) : 0;
        int rw = (win_w == NWH - 1) ? (cm < 4 ? 1 : 2) : 0;
        regm = rh * 3 + rw;
      }
      float vals[4];
      float mx = -3.0e38f;
#pragma unroll
      for (int j = 0; j < 4; ++j) {
        int n  = 16 * j + nl;
        int rn = n >> 3, cn = n & 7;
        float bb = rpbs[((rm - rn + 7) * 15 + (cm - cn + 7)) * 8 + head];
        float v = s[i][j][r] * scale + bb;
        if (masked) {
          int rh = (win_h == NWH - 1) ? (rn < 4 ? 1 : 2) : 0;
          int rw = (win_w == NWH - 1) ? (cn < 4 ? 1 : 2) : 0;
          if (rh * 3 + rw != regm) v -= 100.0f;
        }
        vals[j] = v;
        mx = fmaxf(mx, v);
      }
#pragma unroll
      for (int d = 8; d >= 1; d >>= 1) mx = fmaxf(mx, __shfl_xor(mx, d));
      float sum = 0.0f;
#pragma unroll
      for (int j = 0; j < 4; ++j) { vals[j] = __expf(vals[j] - mx); sum += vals[j]; }
#pragma unroll
      for (int d = 8; d >= 1; d >>= 1) sum += __shfl_xor(sum, d);
      float inv = 1.0f / sum;
#pragma unroll
      for (int j = 0; j < 4; ++j) ps[wv][m][16 * j + nl] = f2bf(vals[j] * inv);
    }
  }
  __syncthreads();

  // ---- out = P(64x64) * V(64x32): 4x2 fragments, 2 K-steps of 32 ----
  v8f o[4][2];
#pragma unroll
  for (int i = 0; i < 4; ++i)
#pragma unroll
    for (int jj = 0; jj < 2; ++jj) o[i][jj] = zero8();
#pragma unroll
  for (int ks = 0; ks < 2; ++ks) {
    v16bf ap[4], bv[2];
#pragma unroll
    for (int i = 0; i < 4; ++i)
      ap[i] = load_a_frag(&ps[wv][16 * i][32 * ks], 64);
#pragma unroll
    for (int jj = 0; jj < 2; ++jj)
      bv[jj] = load_b_frag(&vt[wv][16 * jj][32 * ks], 64);
#pragma unroll
    for (int i = 0; i < 4; ++i)
#pragma unroll
      for (int jj = 0; jj < 2; ++jj)
        o[i][jj] = wmma_bf16(ap[i], bv[jj], o[i][jj]);
  }

  // scatter back to token layout, head-major columns
#pragma unroll
  for (int i = 0; i < 4; ++i) {
#pragma unroll
    for (int jj = 0; jj < 2; ++jj) {
#pragma unroll
      for (int r = 0; r < 8; ++r) {
        int m = 16 * i + r + 8 * g;
        int n = 16 * jj + nl;
        outp[(size_t)toks[m] * CDIM + head * 32 + n] = f2bf(o[i][jj][r]);
      }
    }
  }
}

// ---------------------------------------------------------------------------
// LayerNorm over C=256, one wave per row (residual already added upstream).
// ---------------------------------------------------------------------------
__global__ __launch_bounds__(256) void k_ln(
    const bf16* __restrict__ in, const float* __restrict__ gamma,
    const float* __restrict__ beta, bf16* __restrict__ outp)
{
  int row  = blockIdx.x * 8 + (threadIdx.x >> 5);
  int lane = threadIdx.x & 31;
  const bf16* r = in + (size_t)row * CDIM;
  float v[8];
#pragma unroll
  for (int e = 0; e < 8; ++e) v[e] = bf2f(r[lane + 32 * e]);
  float s = 0.0f;
#pragma unroll
  for (int e = 0; e < 8; ++e) s += v[e];
#pragma unroll
  for (int d = 16; d >= 1; d >>= 1) s += __shfl_xor(s, d);
  float mu = s * (1.0f / 256.0f);
  float q = 0.0f;
#pragma unroll
  for (int e = 0; e < 8; ++e) { float dd = v[e] - mu; q += dd * dd; }
#pragma unroll
  for (int d = 16; d >= 1; d >>= 1) q += __shfl_xor(q, d);
  float rs = rsqrtf(q * (1.0f / 256.0f) + 1e-5f);
  bf16* w = outp + (size_t)row * CDIM;
#pragma unroll
  for (int e = 0; e < 8; ++e) {
    int c = lane + 32 * e;
    w[c] = f2bf((v[e] - mu) * rs * gamma[c] + beta[c]);
  }
}

// ---------------------------------------------------------------------------
// Host orchestration
// ---------------------------------------------------------------------------
extern "C" void kernel_launch(void* const* d_in, const int* in_sizes, int n_in,
                              void* d_out, int out_size, void* d_ws, size_t ws_size,
                              hipStream_t stream)
{
  (void)in_sizes; (void)n_in; (void)out_size; (void)ws_size;
  constexpr int Tn = TTOK;

  char* pws = (char*)d_ws;
  auto alloc = [&](size_t bytes) -> void* {
    void* q = (void*)pws;
    pws += (bytes + 255) & ~(size_t)255;
    return q;
  };
  bf16* tokx = (bf16*)alloc((size_t)Tn * 256 * 2);
  bf16* toky = (bf16*)alloc((size_t)Tn * 256 * 2);
  bf16* qkvx = (bf16*)alloc((size_t)Tn * 768 * 2);
  bf16* qkvy = (bf16*)alloc((size_t)Tn * 768 * 2);
  bf16* ffb  = (bf16*)alloc((size_t)Tn * 512 * 2);
  bf16* tmp  = (bf16*)alloc((size_t)Tn * 256 * 2);
  bf16* att  = (bf16*)alloc((size_t)Tn * 256 * 2);

  auto F = [&](int i) -> const float* { return (const float*)d_in[i]; };
  auto cvtw = [&](int idx, int n) -> bf16* {
    bf16* dst = (bf16*)alloc((size_t)n * 2);
    k_cvt<<<dim3((n + 1023) / 1024), 256, 0, stream>>>(F(idx), dst, n);
    return dst;
  };

  // d_in layout (setup_inputs insertion order): 0=x, 1=y, then
  // sa1..sa4 (wqkv,bqkv,rpb,wproj,bproj,g,b), ca1..ca4
  // (wq,bq,wk,bk,wv,bv,rpb,wproj,bproj,g,b), ff1..ff8 (w1,b1,w2,b2,g,b).
  const int SA0 = 2, CA0 = 2 + 4 * 7, FF0 = 2 + 4 * 7 + 4 * 11;
  auto saI = [&](int i) { return SA0 + 7 * i; };
  auto caI = [&](int i) { return CA0 + 11 * i; };
  auto ffI = [&](int i) { return FF0 + 6 * i; };

  bf16 *sa_wqkv[4], *sa_wpr[4], *ca_wq[4], *ca_wk[4], *ca_wv[4], *ca_wpr[4];
  bf16 *ff_w1[8], *ff_w2[8];
  for (int i = 0; i < 4; ++i) {
    sa_wqkv[i] = cvtw(saI(i) + 0, 768 * 256);
    sa_wpr [i] = cvtw(saI(i) + 3, 256 * 256);
    ca_wq  [i] = cvtw(caI(i) + 0, 256 * 256);
    ca_wk  [i] = cvtw(caI(i) + 2, 256 * 256);
    ca_wv  [i] = cvtw(caI(i) + 4, 256 * 256);
    ca_wpr [i] = cvtw(caI(i) + 7, 256 * 256);
  }
  for (int i = 0; i < 8; ++i) {
    ff_w1[i] = cvtw(ffI(i) + 0, 512 * 256);
    ff_w2[i] = cvtw(ffI(i) + 2, 256 * 512);
  }

  k_to_tok<<<dim3(Tn), 256, 0, stream>>>(F(0), tokx);
  k_to_tok<<<dim3(Tn), 256, 0, stream>>>(F(1), toky);

  auto gemm = [&](const bf16* A, int K, const bf16* W, const float* bias,
                  bf16* outp, int ldOut, int colOff, int Ntot,
                  const bf16* resid, int epi) {
    dim3 grid(Tn / 64, Ntot / 256);
    if (epi == EPI_STORE)
      k_gemm<EPI_STORE><<<grid, 256, 0, stream>>>(A, W, bias, outp, resid, K, ldOut, colOff);
    else if (epi == EPI_GELU)
      k_gemm<EPI_GELU><<<grid, 256, 0, stream>>>(A, W, bias, outp, resid, K, ldOut, colOff);
    else
      k_gemm<EPI_RESID><<<grid, 256, 0, stream>>>(A, W, bias, outp, resid, K, ldOut, colOff);
  };
  auto lnk = [&](const bf16* in, const float* g, const float* b, bf16* o) {
    k_ln<<<dim3(Tn / 8), 256, 0, stream>>>(in, g, b, o);
  };
  auto attn = [&](const bf16* qkv, const float* rpb, int off, int masked) {
    k_attn<<<dim3(NWIN, 4), 64, 0, stream>>>(qkv, qkv, rpb, att, off, masked);
  };
  auto ffn = [&](bf16* tok, int fi) {
    gemm(tok, 256, ff_w1[fi], F(ffI(fi) + 1), ffb, 512, 0, 512, nullptr, EPI_GELU);
    gemm(ffb, 512, ff_w2[fi], F(ffI(fi) + 3), tmp, 256, 0, 256, tok, EPI_RESID);
    lnk(tmp, F(ffI(fi) + 4), F(ffI(fi) + 5), tok);
  };
  auto self_stage = [&](bf16* tok, bf16* qkv, int si, int fi, int off, int masked) {
    gemm(tok, 256, sa_wqkv[si], F(saI(si) + 1), qkv, 768, 0, 768, nullptr, EPI_STORE);
    attn(qkv, F(saI(si) + 2), off, masked);
    gemm(att, 256, sa_wpr[si], F(saI(si) + 4), tmp, 256, 0, 256, tok, EPI_RESID);
    lnk(tmp, F(saI(si) + 5), F(saI(si) + 6), tok);
    ffn(tok, fi);
  };
  auto cross_qkv = [&](bf16* qtok, bf16* kvtok, bf16* qkv, int ci) {
    gemm(qtok,  256, ca_wq[ci], F(caI(ci) + 1), qkv, 768, 0,   256, nullptr, EPI_STORE);
    gemm(kvtok, 256, ca_wk[ci], F(caI(ci) + 3), qkv, 768, 256, 256, nullptr, EPI_STORE);
    gemm(kvtok, 256, ca_wv[ci], F(caI(ci) + 5), qkv, 768, 512, 256, nullptr, EPI_STORE);
  };
  auto cross_tail = [&](bf16* tok, bf16* qkv, int ci, int fi, int off, int masked) {
    attn(qkv, F(caI(ci) + 6), off, masked);
    gemm(att, 256, ca_wpr[ci], F(caI(ci) + 8), tmp, 256, 0, 256, tok, EPI_RESID);
    lnk(tmp, F(caI(ci) + 9), F(caI(ci) + 10), tok);
    ffn(tok, fi);
  };

  // stage 1: unshifted self-attn
  self_stage(tokx, qkvx, 0, 0, 0, 0);
  self_stage(toky, qkvy, 1, 1, 0, 0);
  // stage 2: shifted self-attn (roll -4 => gather offset +4, Swin mask)
  self_stage(tokx, qkvx, 2, 2, 4, 1);
  self_stage(toky, qkvy, 3, 3, 4, 1);
  // stage 3: unshifted cross-attn (q from self stream, k/v from other)
  cross_qkv(tokx, toky, qkvx, 0);
  cross_qkv(toky, tokx, qkvy, 1);
  cross_tail(tokx, qkvx, 0, 4, 0, 0);
  cross_tail(toky, qkvy, 1, 5, 0, 0);
  // stage 4: shifted cross-attn
  cross_qkv(tokx, toky, qkvx, 2);
  cross_qkv(toky, tokx, qkvy, 3);
  cross_tail(tokx, qkvx, 2, 6, 4, 1);
  cross_tail(toky, qkvy, 3, 7, 4, 1);

  // outputs: (xm, ym) concatenated, NCHW f32
  k_from_tok<<<dim3(Tn), 256, 0, stream>>>(tokx, (float*)d_out);
  k_from_tok<<<dim3(Tn), 256, 0, stream>>>(toky, (float*)d_out + (size_t)Tn * 256);
}
